// HeadAffinityGAT_15805479649709
// MI455X (gfx1250) — matrix operand order repaired and checked by reading, not compiled, not run on hardware
//
#include <hip/hip_runtime.h>
#include <hip/hip_bf16.h>
#include <math.h>

#define Bdim   64
#define Ndim   256
#define Ddim   128
#define Kheads 8
#define TWO_D  256

typedef __attribute__((ext_vector_type(16))) _Float16 v16h;
typedef __attribute__((ext_vector_type(8)))  float    v8f;
typedef _Float16 h2 __attribute__((ext_vector_type(2)));
typedef _Float16 h8 __attribute__((ext_vector_type(8)));

__device__ __forceinline__ v8f zero8() {
  v8f z;
#pragma unroll
  for (int i = 0; i < 8; ++i) z[i] = 0.f;
  return z;
}

__device__ __forceinline__ float fast_tanh(float x) {
#if defined(__has_builtin)
# if __has_builtin(__builtin_amdgcn_tanhf)
  return __builtin_amdgcn_tanhf(x);
# elif __has_builtin(__builtin_amdgcn_tanh_f32)
  return __builtin_amdgcn_tanh_f32(x);
# else
  return tanhf(x);
# endif
#else
  return tanhf(x);
#endif
}

// A fragment (16x32 f16, M x K): lane = row (M = lane&15),
// lanes 0-15 hold K {0..7, 16..23}, lanes 16-31 hold K {8..15, 24..31}.
__device__ __forceinline__ v16h frag_a(const _Float16* src, int ld) {
  const int lane = threadIdx.x & 31;
  const _Float16* p = src + (lane & 15) * ld + ((lane >> 4) << 3);
  v16h f;
  const h2* q  = (const h2*)p;        // K = kh .. kh+7
  const h2* q2 = (const h2*)(p + 16); // K = kh+16 .. kh+23
#pragma unroll
  for (int v = 0; v < 4; ++v) {
    h2 a = q[v];  f[2*v]     = a[0]; f[2*v+1]     = a[1];
    h2 b = q2[v]; f[8 + 2*v] = b[0]; f[8 + 2*v+1] = b[1];
  }
  return f;
}

// B fragment (32x16 f16, K x N), read from N-major storage (row n holds K
// contiguously): lane = column (N = lane&15), lanes 0-15 hold K {0..7, 16..23},
// lanes 16-31 hold K {8..15, 24..31}.
__device__ __forceinline__ v16h frag_b(const _Float16* src, int ld) {
  const int lane = threadIdx.x & 31;
  const _Float16* p = src + (lane & 15) * ld + ((lane >> 4) << 3);
  v16h f;
  const h2* q = (const h2*)p;
  const h2* q2 = (const h2*)(p + 16);
#pragma unroll
  for (int v = 0; v < 4; ++v) {
    h2 a = q[v];  f[2*v]     = a[0]; f[2*v+1]     = a[1];
    h2 b = q2[v]; f[8 + 2*v] = b[0]; f[8 + 2*v+1] = b[1];
  }
  return f;
}

// -------- Kernel 1: fused = concat(desc,nv) @ fusion_w^T + fusion_b  (f16 out)
// Each block: 64 rows x 128 cols, K = 256 streamed in 4 chunks of 64.
__global__ void __launch_bounds__(256) fuse_kernel(
    const float* __restrict__ desc, const float* __restrict__ nv,
    const float* __restrict__ fw,   const float* __restrict__ fb,
    _Float16* __restrict__ fusedF16)
{
  __shared__ _Float16 aLds[64 * 64];   //  8 KB
  __shared__ _Float16 wLds[128 * 64];  // 16 KB (also output staging, 64x128)
  const int t    = threadIdx.x;
  const int lane = t & 31, wave = t >> 5;
  const int r0   = blockIdx.x * 64;

  v8f acc[4];
#pragma unroll
  for (int i = 0; i < 4; ++i) acc[i] = zero8();

  for (int ec = 0; ec < 4; ++ec) {
    const int eoff = ec * 64;
    const float* asrc = (eoff < Ddim) ? (desc + (size_t)r0 * Ddim + eoff)
                                      : (nv   + (size_t)r0 * Ddim + (eoff - Ddim));
    // A chunk: 64 rows x 64 e-cols, f32 -> f16, paired b32 LDS stores
#pragma unroll
    for (int it = 0; it < 8; ++it) {
      int p = t + it * 256;             // pair index, 2048 pairs
      int i = p >> 5, j = (p & 31) * 2;
      h2 val;
      val[0] = (_Float16)asrc[(size_t)i * Ddim + j];
      val[1] = (_Float16)asrc[(size_t)i * Ddim + j + 1];
      *(h2*)&aLds[i * 64 + j] = val;
    }
    // B chunk: fusion_w rows d (=output col), e-chunk contiguous, paired
#pragma unroll
    for (int it = 0; it < 16; ++it) {
      int p = t + it * 256;             // 4096 pairs
      int d = p >> 5, j = (p & 31) * 2;
      h2 val;
      val[0] = (_Float16)fw[(size_t)d * TWO_D + eoff + j];
      val[1] = (_Float16)fw[(size_t)d * TWO_D + eoff + j + 1];
      *(h2*)&wLds[d * 64 + j] = val;
    }
    __syncthreads();
#pragma unroll
    for (int kk = 0; kk < 2; ++kk) {
      v16h bf = frag_b(&wLds[(wave * 16) * 64 + kk * 32], 64);
#pragma unroll
      for (int i = 0; i < 4; ++i) {
        v16h af = frag_a(&aLds[(i * 16) * 64 + kk * 32], 64);
        acc[i] = __builtin_amdgcn_wmma_f32_16x16x32_f16(
            false, af, false, bf, (short)0, acc[i], false, false);
      }
    }
    __syncthreads();
  }
  // epilogue: add bias, stage f16 tile (64x128) in wLds, then b128 stores
  const int cl  = lane & 15;
  const int col = wave * 16 + cl;
  const int hi  = (lane >> 4) << 3;
  const float bias = fb[col];
#pragma unroll
  for (int i = 0; i < 4; ++i) {
#pragma unroll
    for (int v = 0; v < 8; ++v)
      wLds[(i * 16 + v + hi) * 128 + col] = (_Float16)(acc[i][v] + bias);
  }
  __syncthreads();
  {
    const uint4* src = (const uint4*)wLds;                       // 1024 uint4
    uint4* dst = (uint4*)(fusedF16 + (size_t)r0 * Ddim);
#pragma unroll
    for (int it = 0; it < 4; ++it) dst[t + it * 256] = src[t + it * 256];
  }
}

// -------- Kernel 2: WsymT[k][f][d] = 0.5*(W[k,d,f]+W[k,f,d]) (f16); uv = 0.5*(ul+ur)
__global__ void prep_kernel(const float* __restrict__ W,
                            const float* __restrict__ ul, const float* __restrict__ ur,
                            _Float16* __restrict__ wsymT, float* __restrict__ uv)
{
  int t = blockIdx.x * 256 + threadIdx.x;   // K*D*D = 131072 threads
  int k = t >> 14;
  int rem = t & 16383;
  int f = rem >> 7, d = rem & 127;
  const float* Wk = W + ((size_t)k << 14);
  wsymT[t] = (_Float16)(0.5f * (Wk[d * Ddim + f] + Wk[f * Ddim + d]));
  if (t < Kheads * Ddim) uv[t] = 0.5f * (ul[t] + ur[t]);
}

// -------- Kernel 3: s[b,k,n] = sum_d fused[b,n,d] * uv[k,d]   (0.5 folded into uv)
__global__ void s_kernel(const _Float16* __restrict__ fusedF16,
                         const float* __restrict__ uv,
                         float* __restrict__ s)
{
  int t = blockIdx.x * 256 + threadIdx.x;   // B*K*N = 131072
  int b = t >> 11;
  int k = (t >> 8) & 7;
  int n = t & 255;
  const h8* fr = (const h8*)(fusedF16 + ((size_t)(b * Ndim + n)) * Ddim);
  const float* u = uv + k * Ddim;
  float acc = 0.f;
#pragma unroll
  for (int d8 = 0; d8 < Ddim / 8; ++d8) {
    h8 v = fr[d8];
#pragma unroll
    for (int j = 0; j < 8; ++j) acc += (float)v[j] * u[d8 * 8 + j];
  }
  s[t] = acc;
}

// -------- Kernel 4: per (b,k,n-strip of 32): z = F_strip @ Wsym (LDS-only),
// out[n,m] = tanh(z[n,:]·F[m,:] + s[n] + s[m] + b[k])
__global__ void __launch_bounds__(256) affinity_kernel(
    const _Float16* __restrict__ fusedF16,
    const _Float16* __restrict__ wsymT,
    const float* __restrict__ s,
    const float* __restrict__ bvec,
    float* __restrict__ out)
{
  __shared__ _Float16 fStrip[32 * 128];   //  8 KB (F strip, then per-wave store staging)
  __shared__ _Float16 wbuf[128 * 128];    // 32 KB (Wsym, reused for F m-chunks)
  __shared__ _Float16 zbuf[32 * 128];     //  8 KB
  __shared__ float    sLds[256];          //  1 KB

  const int t    = threadIdx.x;
  const int lane = t & 31, wave = t >> 5;
  const int id    = blockIdx.x;           // B*K*8 = 4096 blocks
  const int strip = id & 7;
  const int k     = (id >> 3) & 7;
  const int b     = id >> 6;
  const int n0    = strip * 32;

  { // F strip: 32x128 halves = 512 uint4
    const uint4* src = (const uint4*)(fusedF16 + ((size_t)(b * Ndim + n0)) * Ddim);
    uint4* dst = (uint4*)fStrip;
#pragma unroll
    for (int it = 0; it < 2; ++it) dst[t + it * 256] = src[t + it * 256];
  }
  { // WsymT[k]: 128x128 halves = 2048 uint4
    const uint4* src = (const uint4*)(wsymT + ((size_t)k << 14));
    uint4* dst = (uint4*)wbuf;
#pragma unroll
    for (int it = 0; it < 8; ++it) dst[t + it * 256] = src[t + it * 256];
  }
  sLds[t] = s[((size_t)(b * Kheads + k)) * Ndim + t];
  __syncthreads();

  const int hi = (lane >> 4) << 3;
  const int cl = lane & 15;

  // Phase B: z = fStrip(32x128) @ Wsym(128x128) -> zbuf, 2 tiles per wave
#pragma unroll
  for (int half = 0; half < 2; ++half) {
    const int tn = half, tf = wave;
    v8f acc = zero8();
#pragma unroll
    for (int kk = 0; kk < 4; ++kk) {
      v16h af = frag_a(&fStrip[(tn * 16) * 128 + kk * 32], 128);
      v16h bf = frag_b(&wbuf[(tf * 16) * 128 + kk * 32], 128);
      acc = __builtin_amdgcn_wmma_f32_16x16x32_f16(
          false, af, false, bf, (short)0, acc, false, false);
    }
#pragma unroll
    for (int v = 0; v < 8; ++v)
      zbuf[(tn * 16 + v + hi) * 128 + tf * 16 + cl] = (_Float16)acc[v];
  }
  __syncthreads();   // also: fStrip reads done -> safe to reuse as staging

  const float bk = bvec[k];
  float* stage = ((float*)fStrip) + wave * 256;   // per-wave 16x16 f32 tile

  // Phase C: out strip = z(32x128) @ F^T, m in chunks of 64, fused epilogue
  for (int mc = 0; mc < 4; ++mc) {
    { // F m-chunk: 64x128 halves = 1024 uint4, overlays wbuf
      const uint4* src = (const uint4*)(fusedF16 + ((size_t)(b * Ndim + mc * 64)) * Ddim);
      uint4* dst = (uint4*)wbuf;
#pragma unroll
      for (int it = 0; it < 4; ++it) dst[t + it * 256] = src[t + it * 256];
    }
    __syncthreads();

    const int tn = wave >> 2, tm = wave & 3;
    v8f acc = zero8();
#pragma unroll
    for (int kk = 0; kk < 4; ++kk) {
      v16h af = frag_a(&zbuf[(tn * 16) * 128 + kk * 32], 128);
      v16h bf = frag_b(&wbuf[(tm * 16) * 128 + kk * 32], 128);
      acc = __builtin_amdgcn_wmma_f32_16x16x32_f16(
          false, af, false, bf, (short)0, acc, false, false);
    }
    // epilogue: tanh(acc + s_n + s_m + b_k); round-trip the 16x16 tile
    // through per-wave LDS staging so global stores are contiguous b128.
    const int m  = mc * 64 + tm * 16 + cl;
    const float sm = sLds[m];
#pragma unroll
    for (int v = 0; v < 8; ++v) {
      int n = n0 + tn * 16 + hi + v;
      stage[(hi + v) * 16 + cl] = fast_tanh(acc[v] + sLds[n] + sm + bk);
    }
    // LDS ops are in-order within a wave: read back rows, store b128 pairs.
    {
      const int r = lane >> 1;            // tile row 0..15
      const int c = (lane & 1) * 8;       // col offset 0 or 8
      float4 w0 = *(const float4*)&stage[r * 16 + c];
      float4 w1 = *(const float4*)&stage[r * 16 + c + 4];
      const int n  = n0 + tn * 16 + r;
      const int mg = mc * 64 + tm * 16 + c;
      float* orow = out + ((size_t)(b * Kheads + k)) * Ndim * Ndim
                        + (size_t)n * Ndim + mg;
      *(float4*)orow       = w0;
      *(float4*)(orow + 4) = w1;
    }
    __syncthreads();
  }
}

extern "C" void kernel_launch(void* const* d_in, const int* in_sizes, int n_in,
                              void* d_out, int out_size, void* d_ws, size_t ws_size,
                              hipStream_t stream)
{
  (void)in_sizes; (void)n_in; (void)out_size; (void)ws_size;
  const float* desc = (const float*)d_in[0];
  const float* nv   = (const float*)d_in[1];
  const float* fw   = (const float*)d_in[2];
  const float* fb   = (const float*)d_in[3];
  const float* W    = (const float*)d_in[4];
  const float* ul   = (const float*)d_in[5];
  const float* ur   = (const float*)d_in[6];
  const float* bv   = (const float*)d_in[7];
  float* out = (float*)d_out;

  char* ws = (char*)d_ws;
  _Float16* fusedF16 = (_Float16*)ws;                              // 4 MB
  _Float16* wsymT    = (_Float16*)(ws + (4u << 20));               // 256 KB
  float*    uv       = (float*)(ws + (4u << 20) + (256u << 10));   // 4 KB
  float*    sbuf     = (float*)(ws + (4u << 20) + (260u << 10));   // 512 KB

  fuse_kernel    <<<256, 256, 0, stream>>>(desc, nv, fw, fb, fusedF16);
  prep_kernel    <<<512, 256, 0, stream>>>(W, ul, ur, wsymT, uv);
  s_kernel       <<<512, 256, 0, stream>>>(fusedF16, uv, sbuf);
  affinity_kernel<<<4096, 256, 0, stream>>>(fusedF16, wsymT, sbuf, bv, out);
}
